// WaveNetModel_44281112822295
// MI455X (gfx1250) — matrix operand order repaired and checked by reading, not compiled, not run on hardware
//
#include <hip/hip_runtime.h>
#include <hip/hip_bf16.h>

typedef _Float16 f16;
typedef _Float16 h8  __attribute__((ext_vector_type(8)));
typedef _Float16 v16h __attribute__((ext_vector_type(16)));
typedef float    v8f  __attribute__((ext_vector_type(8)));

#define BB   4
#define TT   16000
#define RR   256
#define SS   512
#define OO   256
#define LL   16
#define DILATION 256
#define MROWS (BB*TT)   // 64000

// ---------------------------------------------------------------------------
// Pack an (N_out x K_in) f32 weight matrix into f16 WMMA B-fragments.
// Fragment (kt,nt) holds B[k = kt*32 + (lane>>4)*8 + (h&7) + (h>>3)*16][n = nt*16 + (lane&15)]
// at dst[((kt*NT + nt)*32 + lane)*16 + h]  (32B contiguous per lane).
// ---------------------------------------------------------------------------
__global__ void pack_w16(const float* __restrict__ src, f16* __restrict__ dst,
                         int K, int N, int strideN, int strideK) {
  int idx = blockIdx.x * blockDim.x + threadIdx.x;
  if (idx >= K * N) return;
  int h    = idx & 15;
  int lane = (idx >> 4) & 31;
  int fidx = idx >> 9;
  int NT   = N >> 4;
  int nt   = fidx % NT;
  int kt   = fidx / NT;
  int n = nt * 16 + (lane & 15);
  int k = kt * 32 + (lane >> 4) * 8 + (h & 7) + (h >> 3) * 16;
  dst[idx] = (f16)src[(size_t)n * strideN + (size_t)k * strideK];
}

__global__ void embed_k(const int* __restrict__ tokens,
                        const float* __restrict__ embed,
                        f16* __restrict__ x) {
  int idx = blockIdx.x * 256 + threadIdx.x;   // over MROWS*256
  int row = idx >> 8;
  int c   = idx & 255;
  x[idx] = (f16)embed[(size_t)tokens[row] * RR + c];
}

// ---------------------------------------------------------------------------
// A-fragment (16x32 f16) load: two b128 chunks per documented CDNA5 layout.
// ---------------------------------------------------------------------------
__device__ __forceinline__ v16h load_a(const f16* __restrict__ p) {
  v16h a;
  ((h8*)&a)[0] = *(const h8*)(p);
  ((h8*)&a)[1] = *(const h8*)(p + 16);
  return a;
}

// ---------------------------------------------------------------------------
// Software-pipelined 16x(16*4) GEMM tile: NKS k-steps of 32, 4 n-tiles.
// aBase = &ldsA[aRow*rowStride + aKoff] (advance 32 per k-step),
// wLane = W + lane*16; fragment (ks,nt) at wLane + (ks*NT + nt + ntBase)*512.
// Loads for step k+1 are issued before the 4 WMMAs of step k.
// ---------------------------------------------------------------------------
template<int NKS>
__device__ __forceinline__ void gemm_tile4(const f16* __restrict__ aBase,
                                           const f16* __restrict__ wLane,
                                           int NT, int ntBase, v8f acc[4]) {
  v16h a0 = load_a(aBase);
  v16h b0[4];
#pragma unroll
  for (int nt = 0; nt < 4; ++nt)
    b0[nt] = *(const v16h*)(wLane + (size_t)(ntBase + nt) * 512);
#pragma unroll
  for (int ks = 0; ks < NKS; ++ks) {
    v16h a1 = a0;
    v16h b1[4] = {b0[0], b0[1], b0[2], b0[3]};
    if (ks + 1 < NKS) {
      a1 = load_a(aBase + (ks + 1) * 32);
#pragma unroll
      for (int nt = 0; nt < 4; ++nt)
        b1[nt] = *(const v16h*)(wLane + (size_t)((ks + 1) * NT + ntBase + nt) * 512);
    }
#pragma unroll
    for (int nt = 0; nt < 4; ++nt)
      acc[nt] = __builtin_amdgcn_wmma_f32_16x16x32_f16(
          false, a0, false, b0[nt], (short)0, acc[nt], false, false);
    a0 = a1;
#pragma unroll
    for (int nt = 0; nt < 4; ++nt) b0[nt] = b1[nt];
  }
}

// ---------------------------------------------------------------------------
// One fused WaveNet layer for a 32-row (b,t) tile.
//   g     = Wg0*x[t-256] + Wg1*x[t] + b   (only the gated 256 rows)
//   act   = tanh(g)*sigmoid(g)            (LDS f16)
//   xnext = Wres*act + res_b + x          (staged in LDS, coalesced writeback)
//   skip += Wskip*act + skip_b            (f16 tile accumulated in LDS)
// Block = 256 threads = 8 waves arranged 2(M) x 4(N).
// ---------------------------------------------------------------------------
__global__ __launch_bounds__(256) void layer_k(
    const f16* __restrict__ xcur, f16* __restrict__ xnext,
    f16* __restrict__ skipacc,
    const f16* __restrict__ Wg0, const f16* __restrict__ Wg1,
    const f16* __restrict__ Wres, const f16* __restrict__ Wskip,
    const float* __restrict__ dilate_b, const float* __restrict__ res_b,
    const float* __restrict__ skip_b,
    int hasRes, int first) {
  __shared__ f16 ldsX[32 * 256];    // x tile; later reused as xnext staging
  __shared__ f16 ldsXs[32 * 256];   // x shifted tile
  __shared__ f16 ldsAct[32 * 256];  // gated activation
  __shared__ f16 ldsSk[32 * 512];   // skip accumulator tile

  const int tid    = threadIdx.x;
  const int lane   = tid & 31;
  const int wave   = tid >> 5;
  const int mSub   = wave & 1;
  const int nGroup = wave >> 1;
  const int row0   = blockIdx.x * 32;
  const int t0     = row0 % TT;
  const bool hasShift = (t0 >= DILATION);

  // ---- stage x, x_shift, and current skip tile into LDS (all b128) -------
  for (int i = tid; i < (32 * 256) / 8; i += 256) {
    int e = i * 8;
    int r = e >> 8;
    int c = e & 255;
    int row = row0 + r;
    *(h8*)&ldsX[e] = *(const h8*)&xcur[(size_t)row * 256 + c];
    h8 z;
    if (hasShift) {
      z = *(const h8*)&xcur[(size_t)(row - DILATION) * 256 + c];
    } else {
      for (int j = 0; j < 8; ++j) z[j] = (f16)0.f;
    }
    *(h8*)&ldsXs[e] = z;
  }
  for (int i = tid; i < (32 * 512) / 8; i += 256) {
    int e = i * 8;
    int r = e >> 9;
    int c = e & 511;
    h8 z;
    if (first) {
      for (int j = 0; j < 8; ++j) z[j] = (f16)0.f;
    } else {
      z = *(const h8*)&skipacc[(size_t)(row0 + r) * 512 + c];
    }
    *(h8*)&ldsSk[e] = z;
  }
  // warm L2/L0 for weights used later in this block
  __builtin_prefetch(&Wskip[((size_t)(nGroup * 8) * 32 + lane) * 16], 0, 3);
  __builtin_prefetch(&Wres[((size_t)(nGroup * 4) * 32 + lane) * 16], 0, 3);
  __syncthreads();

  const int nIn   = lane & 15;
  const int mOff  = (lane >> 4) * 8;
  const int aRow  = mSub * 16 + (lane & 15);
  const int aKoff = (lane >> 4) * 8;
  const f16* wLaneG1 = Wg1 + lane * 16;
  const f16* wLaneG0 = Wg0 + lane * 16;
  const f16* wLaneR  = Wres + lane * 16;
  const f16* wLaneS  = Wskip + lane * 16;

  // ---- Phase A: gated-unit GEMM (N=256, K=256 twice) ---------------------
  v8f acc[4];
  for (int nt = 0; nt < 4; ++nt)
    for (int j = 0; j < 8; ++j) acc[nt][j] = 0.f;

  gemm_tile4<8>(&ldsX[aRow * 256 + aKoff], wLaneG1, 16, nGroup * 4, acc);
  if (hasShift)
    gemm_tile4<8>(&ldsXs[aRow * 256 + aKoff], wLaneG0, 16, nGroup * 4, acc);

  // bias + tanh(g)*sigmoid(g) -> ldsAct
#pragma unroll
  for (int nt = 0; nt < 4; ++nt) {
    int n = nGroup * 64 + nt * 16 + nIn;
    float bv = dilate_b[n];
#pragma unroll
    for (int j = 0; j < 8; ++j) {
      float g = acc[nt][j] + bv;
      float sg = 1.f / (1.f + __expf(-g));
      float s2 = 1.f / (1.f + __expf(-2.f * g));
      float act = (2.f * s2 - 1.f) * sg;
      int m = mSub * 16 + mOff + j;
      ldsAct[m * 256 + n] = (f16)act;
    }
  }
  __syncthreads();   // act complete; ldsX free for reuse after this point

  const f16* actBase = &ldsAct[aRow * 256 + aKoff];

  // ---- Phase B1: residual GEMM (N=256, K=256), staged into ldsX ----------
  if (hasRes) {
    v8f r4[4];
    for (int nt = 0; nt < 4; ++nt)
      for (int j = 0; j < 8; ++j) r4[nt][j] = 0.f;
    gemm_tile4<8>(actBase, wLaneR, 16, nGroup * 4, r4);
#pragma unroll
    for (int nt = 0; nt < 4; ++nt) {
      int n = nGroup * 64 + nt * 16 + nIn;
      float bv = res_b[n];
#pragma unroll
      for (int j = 0; j < 8; ++j) {
        int m = mSub * 16 + mOff + j;
        float xv = (float)ldsX[m * 256 + n];
        ldsX[m * 256 + n] = (f16)(r4[nt][j] + bv + xv);
      }
    }
  }

  // ---- Phase B2: skip GEMM (N=512, K=256), accumulate into ldsSk ---------
#pragma unroll
  for (int half = 0; half < 2; ++half) {
    v8f s4[4];
    for (int nt = 0; nt < 4; ++nt)
      for (int j = 0; j < 8; ++j) s4[nt][j] = 0.f;
    gemm_tile4<8>(actBase, wLaneS, 32, nGroup * 8 + half * 4, s4);
#pragma unroll
    for (int nt = 0; nt < 4; ++nt) {
      int n = nGroup * 128 + half * 64 + nt * 16 + nIn;
      float bv = skip_b[n];
#pragma unroll
      for (int j = 0; j < 8; ++j) {
        int m = mSub * 16 + mOff + j;
        float v = (float)ldsSk[m * 512 + n] + s4[nt][j] + bv;
        ldsSk[m * 512 + n] = (f16)v;
      }
    }
  }
  __syncthreads();

  // ---- coalesced writeback: xnext and skip tiles -------------------------
  if (hasRes) {
    for (int i = tid; i < (32 * 256) / 8; i += 256) {
      int e = i * 8;
      int r = e >> 8;
      int c = e & 255;
      *(h8*)&xnext[(size_t)(row0 + r) * 256 + c] = *(const h8*)&ldsX[e];
    }
  }
  for (int i = tid; i < (32 * 512) / 8; i += 256) {
    int e = i * 8;
    int r = e >> 9;
    int c = e & 511;
    *(h8*)&skipacc[(size_t)(row0 + r) * 512 + c] = *(const h8*)&ldsSk[e];
  }
}

// ---------------------------------------------------------------------------
// Output head: relu(skip) -> conv_out (K=512) -> relu -> conv_end (K=256)
// -> softmax over 256 channels, written in (b, p, t) layout.
// ---------------------------------------------------------------------------
__global__ __launch_bounds__(256) void out_k(
    const f16* __restrict__ skipacc,
    const f16* __restrict__ Wout, const f16* __restrict__ Wend,
    float* __restrict__ out) {
  __shared__ f16 ldsS[32 * 512];    // relu(skip) f16; later f32 logits alias
  __shared__ f16 ldsH[32 * 256];    // hidden f16
  __shared__ float rowMax[32];
  __shared__ float rowInv[32];
  float* ldsL = (float*)ldsS;

  const int tid    = threadIdx.x;
  const int lane   = tid & 31;
  const int wave   = tid >> 5;
  const int mSub   = wave & 1;
  const int nGroup = wave >> 1;
  const int row0   = blockIdx.x * 32;

  for (int i = tid; i < (32 * 512) / 8; i += 256) {
    int e = i * 8;
    int r = e >> 9;
    int c = e & 511;
    h8 v = *(const h8*)&skipacc[(size_t)(row0 + r) * 512 + c];
    h8 o;
    for (int j = 0; j < 8; ++j) o[j] = (f16)fmaxf((float)v[j], 0.f);
    *(h8*)&ldsS[e] = o;
  }
  __syncthreads();

  const int nIn   = lane & 15;
  const int mOff  = (lane >> 4) * 8;
  const int aRow  = mSub * 16 + (lane & 15);
  const int aKoff = (lane >> 4) * 8;

  // Phase 1: h1 = relu(skip) * Wout^T   (K=512, N=256)
  v8f h[4];
  for (int nt = 0; nt < 4; ++nt)
    for (int j = 0; j < 8; ++j) h[nt][j] = 0.f;
  gemm_tile4<16>(&ldsS[aRow * 512 + aKoff], Wout + lane * 16, 16, nGroup * 4, h);
#pragma unroll
  for (int nt = 0; nt < 4; ++nt) {
    int n = nGroup * 64 + nt * 16 + nIn;
#pragma unroll
    for (int j = 0; j < 8; ++j) {
      int m = mSub * 16 + mOff + j;
      ldsH[m * 256 + n] = (f16)fmaxf(h[nt][j], 0.f);
    }
  }
  __syncthreads();

  // Phase 2: logits = h1 * Wend^T   (K=256, N=256); write f32 to ldsL
  v8f o4[4];
  for (int nt = 0; nt < 4; ++nt)
    for (int j = 0; j < 8; ++j) o4[nt][j] = 0.f;
  gemm_tile4<8>(&ldsH[aRow * 256 + aKoff], Wend + lane * 16, 16, nGroup * 4, o4);
#pragma unroll
  for (int nt = 0; nt < 4; ++nt) {
    int n = nGroup * 64 + nt * 16 + nIn;
#pragma unroll
    for (int j = 0; j < 8; ++j) {
      int m = mSub * 16 + mOff + j;
      ldsL[m * 256 + n] = o4[nt][j];
    }
  }
  __syncthreads();

  // Softmax over 256 channels per row; 8 lanes cooperate per row.
  {
    int r  = tid >> 3;
    int cg = tid & 7;
    float mx = -3.4e38f;
    for (int c = cg * 32; c < cg * 32 + 32; ++c)
      mx = fmaxf(mx, ldsL[r * 256 + c]);
    for (int off = 1; off < 8; off <<= 1)
      mx = fmaxf(mx, __shfl_xor(mx, off, 8));
    float sum = 0.f;
    for (int c = cg * 32; c < cg * 32 + 32; ++c)
      sum += __expf(ldsL[r * 256 + c] - mx);
    for (int off = 1; off < 8; off <<= 1)
      sum += __shfl_xor(sum, off, 8);
    if (cg == 0) { rowMax[r] = mx; rowInv[r] = 1.f / sum; }
  }
  __syncthreads();

  // Write (b, n, t): thread n = tid, 32 consecutive t, float4 stores.
  {
    int n  = tid;
    int bI = row0 / TT;
    int tL = row0 % TT;
    float* obase = out + ((size_t)bI * OO + n) * TT + tL;
    for (int r0 = 0; r0 < 32; r0 += 4) {
      float4 v;
      v.x = __expf(ldsL[(r0 + 0) * 256 + n] - rowMax[r0 + 0]) * rowInv[r0 + 0];
      v.y = __expf(ldsL[(r0 + 1) * 256 + n] - rowMax[r0 + 1]) * rowInv[r0 + 1];
      v.z = __expf(ldsL[(r0 + 2) * 256 + n] - rowMax[r0 + 2]) * rowInv[r0 + 2];
      v.w = __expf(ldsL[(r0 + 3) * 256 + n] - rowMax[r0 + 3]) * rowInv[r0 + 3];
      *(float4*)&obase[r0] = v;
    }
  }
}

// ---------------------------------------------------------------------------
extern "C" void kernel_launch(void* const* d_in, const int* in_sizes, int n_in,
                              void* d_out, int out_size, void* d_ws, size_t ws_size,
                              hipStream_t stream) {
  const int*   tokens     = (const int*)d_in[0];
  const float* embed      = (const float*)d_in[1];
  const float* dilate_w   = (const float*)d_in[2];
  const float* dilate_b   = (const float*)d_in[3];
  const float* res_w      = (const float*)d_in[4];
  const float* res_b      = (const float*)d_in[5];
  const float* skip_w     = (const float*)d_in[6];
  const float* skip_b     = (const float*)d_in[7];
  const float* conv_out_w = (const float*)d_in[8];
  const float* conv_end_w = (const float*)d_in[9];

  size_t off = 0;
  auto alloc = [&](size_t bytes) {
    size_t cur = off;
    off = (off + bytes + 255) & ~(size_t)255;
    return cur;
  };
  char* ws = (char*)d_ws;
  f16* x0    = (f16*)(ws + alloc((size_t)MROWS * 256 * 2));
  f16* x1    = (f16*)(ws + alloc((size_t)MROWS * 256 * 2));
  f16* skipA = (f16*)(ws + alloc((size_t)MROWS * 512 * 2));
  f16* wg0   = (f16*)(ws + alloc((size_t)LL * 256 * 256 * 2));
  f16* wg1   = (f16*)(ws + alloc((size_t)LL * 256 * 256 * 2));
  f16* wres  = (f16*)(ws + alloc((size_t)(LL - 1) * 256 * 256 * 2));
  f16* wskip = (f16*)(ws + alloc((size_t)LL * 256 * 512 * 2));
  f16* wout  = (f16*)(ws + alloc((size_t)512 * 256 * 2));
  f16* wend  = (f16*)(ws + alloc((size_t)256 * 256 * 2));
  if (off > ws_size) return;

  auto gp = [](int total) { return (total + 255) / 256; };

  for (int l = 0; l < LL; ++l) {
    pack_w16<<<gp(256 * 256), 256, 0, stream>>>(
        dilate_w + (size_t)l * 512 * 256 * 2 + 0, wg0 + (size_t)l * 65536,
        256, 256, 512, 2);
    pack_w16<<<gp(256 * 256), 256, 0, stream>>>(
        dilate_w + (size_t)l * 512 * 256 * 2 + 1, wg1 + (size_t)l * 65536,
        256, 256, 512, 2);
    pack_w16<<<gp(512 * 256), 256, 0, stream>>>(
        skip_w + (size_t)l * 512 * 256, wskip + (size_t)l * 131072,
        256, 512, 256, 1);
  }
  for (int l = 0; l < LL - 1; ++l) {
    pack_w16<<<gp(256 * 256), 256, 0, stream>>>(
        res_w + (size_t)l * 256 * 256, wres + (size_t)l * 65536,
        256, 256, 256, 1);
  }
  pack_w16<<<gp(512 * 256), 256, 0, stream>>>(conv_out_w, wout, 512, 256, 512, 1);
  pack_w16<<<gp(256 * 256), 256, 0, stream>>>(conv_end_w, wend, 256, 256, 256, 1);

  embed_k<<<MROWS, 256, 0, stream>>>(tokens, embed, x0);

  f16* xc = x0;
  f16* xn = x1;
  for (int l = 0; l < LL; ++l) {
    int lr = (l < LL - 1) ? l : (LL - 2);
    layer_k<<<MROWS / 32, 256, 0, stream>>>(
        xc, xn, skipA,
        wg0 + (size_t)l * 65536, wg1 + (size_t)l * 65536,
        wres + (size_t)lr * 65536, wskip + (size_t)l * 131072,
        dilate_b + (size_t)l * 512, res_b + (size_t)lr * 256,
        skip_b + (size_t)l * 512,
        (l < LL - 1) ? 1 : 0, (l == 0) ? 1 : 0);
    if (l < LL - 1) { f16* t = xc; xc = xn; xn = t; }
  }

  out_k<<<MROWS / 32, 256, 0, stream>>>(skipA, wout, wend, (float*)d_out);
}
